// _MultiheadAttention_36112085025148
// MI455X (gfx1250) — matrix operand order repaired and checked by reading, not compiled, not run on hardware
//
#include <hip/hip_runtime.h>
#include <hip/hip_bf16.h>

// ---------------------------------------------------------------------------
// Multi-head attention for MI455X (gfx1250, wave32, WMMA bf16 16x16x32,
// async global->LDS staging for pure-bf16 tiles).
// d_out = [output (8*1024*1024) | attn_weights (8*16*1024*1024) | attn_scores (same)] fp32
// d_ws  = q_ws bf16 16MB | kT_ws bf16 16MB | v_ws bf16 16MB | ctx_ws bf16 16MB
// ---------------------------------------------------------------------------

typedef __attribute__((ext_vector_type(16))) __bf16 v16bf;
typedef __attribute__((ext_vector_type(8)))  float  v8f;

#define N_BS     8
#define N_SEQ    1024
#define N_DM     1024
#define N_HEADS  16
#define N_DK     64
#define ATTN_SCALE 0.125f   // 64^-0.5

union FragU { v16bf v; uint4 q[2]; };

__device__ __forceinline__ v16bf load_frag16(const __bf16* p0, const __bf16* p1) {
  FragU f;
  f.q[0] = *(const uint4*)p0;
  f.q[1] = *(const uint4*)p1;
  return f.v;
}

// A matrix (M x 32, bf16) fragment from LDS, row-major with given element stride.
// lane l: M = mrow + (l&15); elements 0..7 -> K = kcol + half*8 + e,
//         elements 8..15 -> K = kcol + 16 + half*8 + (e-8), half = l>>4.
__device__ __forceinline__ v16bf load_a_frag(const __bf16* base, int stride,
                                             int mrow, int kcol, int lrow, int lhalf) {
  const __bf16* r = base + (mrow + lrow) * stride + kcol + lhalf * 8;
  return load_frag16(r, r + 16);
}

// B matrix (32 x N, bf16) fragment from LDS: lane l -> N = ncol + (l&15),
// elements 0..7 -> K = krow + half*8 + e, 8..15 -> K = krow + 16 + half*8 + (e-8).
__device__ __forceinline__ v16bf load_b_frag(const __bf16* base, int stride,
                                             int krow, int ncol, int lane) {
  const __bf16* r = base + krow * stride + ncol + (lane & 15);
  const int hb = (lane >> 4) * 8;
  v16bf f;
#pragma unroll
  for (int s = 0; s < 16; ++s) {
    int k = (s < 8) ? (hb + s) : (16 + hb + (s - 8));
    f[s] = r[k * stride];
  }
  return f;
}

__device__ __forceinline__ v8f wmma_bf16(v16bf a, v16bf b, v8f c) {
  return __builtin_amdgcn_wmma_f32_16x16x32_bf16(false, a, false, b, (short)0, c,
                                                 false, false);
}

__device__ __forceinline__ void store_pair_bf16(__bf16* dst, float x, float y) {
  union { unsigned int u; __bf16 h[2]; } p;
  p.h[0] = (__bf16)x;
  p.h[1] = (__bf16)y;
  *(unsigned int*)dst = p.u;
}

// ---------------------------------------------------------------------------
// Async global->LDS (CDNA5 GLOBAL_LOAD_ASYNC_TO_LDS_B128, tracked by ASYNCcnt).
// LDS flat addresses carry the LDS offset in the low 32 bits (aperture rule),
// so a generic->as(3) cast + ptrtoint yields the ds byte offset.
// ---------------------------------------------------------------------------
typedef __attribute__((address_space(3))) char lds_char_t;

__device__ __forceinline__ unsigned lds_offset_of(const void* p) {
  return (unsigned)(unsigned long long)(lds_char_t*)p;
}

__device__ __forceinline__ void async_load_lds_b128(const void* gsrc, void* ldst) {
  unsigned loff = lds_offset_of(ldst);
  unsigned long long ga = (unsigned long long)gsrc;
  asm volatile("global_load_async_to_lds_b128 %0, %1, off"
               :
               : "v"(loff), "v"(ga)
               : "memory");
}

__device__ __forceinline__ void wait_async_all() {
  asm volatile("s_wait_asynccnt 0" ::: "memory");
}

// ---------------------------------------------------------------------------
// Kernel 1: Q/K/V projection.  out = (X @ W + bias) -> bf16 workspace.
// X: [8192,1024] fp32, W: [1024,1024] fp32.
// transposed==0: out[(b*16+h)*1024 + s][dk]   (q/v layout [b,h,s,dk])
// transposed==1: out[(b*16+h)*64 + dk][s]     (kT layout [b,h,dk,s])
// Block tile 128x128, BK=32, 8 waves in 2x4, each wave 4x2 WMMA tiles.
// (fp32 sources need VALU conversion, so staging stays on the cvt+ds path.)
// ---------------------------------------------------------------------------
__global__ __launch_bounds__(256)
void mha_proj_qkv_kernel(const float* __restrict__ X, const float* __restrict__ W,
                         const float* __restrict__ bias, __bf16* __restrict__ out,
                         int transposed) {
  __shared__ __bf16 As[128 * 40];   // 128 x 32, stride 40 (80B, conflict-free)
  __shared__ __bf16 Bs[32 * 136];   // 32 x 128, stride 136

  const int tid = threadIdx.x;
  const int lane = tid & 31, lrow = lane & 15, lhalf = lane >> 4;
  const int wid = tid >> 5, wm = wid >> 2, wn = wid & 3;
  const int m0 = blockIdx.x * 128, n0 = blockIdx.y * 128;

  v8f acc[4][2] = {};

  for (int kb = 0; kb < N_DM; kb += 32) {
    // stage A 128x32 (fp32 -> bf16): 2048 float2 over 256 threads
#pragma unroll
    for (int i = 0; i < 8; ++i) {
      int p = tid + i * 256;
      int row = p >> 4, cp = p & 15;
      float2 g = *(const float2*)(X + (size_t)(m0 + row) * N_DM + kb + cp * 2);
      store_pair_bf16(&As[row * 40 + cp * 2], g.x, g.y);
    }
    // stage B 32x128
#pragma unroll
    for (int i = 0; i < 8; ++i) {
      int p = tid + i * 256;
      int row = p >> 6, cp = p & 63;
      float2 g = *(const float2*)(W + (size_t)(kb + row) * N_DM + n0 + cp * 2);
      store_pair_bf16(&Bs[row * 136 + cp * 2], g.x, g.y);
    }
    __syncthreads();

    v16bf af[4], bfr[2];
#pragma unroll
    for (int mt = 0; mt < 4; ++mt)
      af[mt] = load_a_frag(As, 40, wm * 64 + mt * 16, 0, lrow, lhalf);
#pragma unroll
    for (int nt = 0; nt < 2; ++nt)
      bfr[nt] = load_b_frag(Bs, 136, 0, wn * 32 + nt * 16, lane);
#pragma unroll
    for (int mt = 0; mt < 4; ++mt)
#pragma unroll
      for (int nt = 0; nt < 2; ++nt)
        acc[mt][nt] = wmma_bf16(af[mt], bfr[nt], acc[mt][nt]);
    __syncthreads();
  }

#pragma unroll
  for (int mt = 0; mt < 4; ++mt)
#pragma unroll
    for (int nt = 0; nt < 2; ++nt)
#pragma unroll
      for (int r = 0; r < 8; ++r) {
        int m = m0 + wm * 64 + mt * 16 + r + lhalf * 8;
        int n = n0 + wn * 32 + nt * 16 + lrow;
        float v = acc[mt][nt][r] + bias[n];
        int b = m >> 10, s = m & 1023, h = n >> 6, dk = n & 63;
        size_t idx = transposed
                         ? (size_t)((b * N_HEADS + h) * N_DK + dk) * N_SEQ + s
                         : (size_t)((b * N_HEADS + h) * N_SEQ + s) * N_DK + dk;
        out[idx] = (__bf16)v;
      }
}

// ---------------------------------------------------------------------------
// Kernel 2: scores = (q @ kT) * scale, softmax, dump scores + weights.
// One WG per 32-row strip per (b,h). Dynamic LDS:
//   S[32][1024] f32 (128KB) | Kt[64][1032] bf16 (129KB) | Qs[32][72] bf16 | pmax/psum
// Q and kT tiles are pure bf16 copies -> async global->LDS DMA.
// ---------------------------------------------------------------------------
__global__ __launch_bounds__(256)
void mha_attn_softmax_kernel(const __bf16* __restrict__ q_ws,
                             const __bf16* __restrict__ kT_ws,
                             float* __restrict__ scores,
                             float* __restrict__ weights) {
  extern __shared__ char smem[];
  float* S = (float*)smem;                                         // 32*1024 f32
  __bf16* Kt = (__bf16*)(smem + 32 * 1024 * 4);                    // 64*1032 bf16
  __bf16* Qs = (__bf16*)(smem + 32 * 1024 * 4 + 64 * 1032 * 2);    // 32*72 bf16
  float* pmax = (float*)(smem + 32 * 1024 * 4 + 64 * 1032 * 2 + 32 * 72 * 2);
  float* psum = pmax + 32 * 8;

  const int tid = threadIdx.x;
  const int lane = tid & 31, lrow = lane & 15, lhalf = lane >> 4;
  const int wid = tid >> 5;
  const int bh = blockIdx.y;
  const int m0 = blockIdx.x * 32;

  // stage Q strip 32x64 bf16: 256 x 16B async copies (1 per thread)
  {
    const __bf16* src = q_ws + ((size_t)bh * N_SEQ + m0) * N_DK;
    int row = tid >> 3, c4 = tid & 7;
    async_load_lds_b128(src + row * N_DK + c4 * 8, &Qs[row * 72 + c4 * 8]);
  }
  // stage full kT 64x1024 bf16 (128KB): 8192 x 16B async copies (32 per thread)
  {
    const __bf16* src = kT_ws + (size_t)bh * N_DK * N_SEQ;
#pragma unroll
    for (int i = 0; i < 32; ++i) {
      int p = tid + i * 256;
      int row = p >> 7, c4 = p & 127;
      async_load_lds_b128(src + row * N_SEQ + c4 * 8, &Kt[row * 1032 + c4 * 8]);
    }
  }
  wait_async_all();
  __syncthreads();

  // GEMM: 32 x 1024 = q(32x64) @ kT(64x1024); wave wid owns n = wid*128..+127
  v16bf af[2][2];
#pragma unroll
  for (int mt = 0; mt < 2; ++mt)
#pragma unroll
    for (int kk = 0; kk < 2; ++kk)
      af[mt][kk] = load_a_frag(Qs, 72, mt * 16, kk * 32, lrow, lhalf);

#pragma unroll
  for (int nt = 0; nt < 8; ++nt) {
    int nb = wid * 128 + nt * 16;
    v8f a0 = {}, a1 = {};
#pragma unroll
    for (int kk = 0; kk < 2; ++kk) {
      v16bf bb = load_b_frag(Kt, 1032, kk * 32, nb, lane);
      a0 = wmma_bf16(af[0][kk], bb, a0);
      a1 = wmma_bf16(af[1][kk], bb, a1);
    }
    int n = nb + lrow;
#pragma unroll
    for (int r = 0; r < 8; ++r) {
      S[(r + lhalf * 8) * 1024 + n] = a0[r] * ATTN_SCALE;
      S[(16 + r + lhalf * 8) * 1024 + n] = a1[r] * ATTN_SCALE;
    }
  }
  __syncthreads();

  // dump raw scores (coalesced 16B)
  {
    uint4* dst = (uint4*)(scores + ((size_t)bh * N_SEQ + m0) * N_SEQ);
    const uint4* s4 = (const uint4*)S;
#pragma unroll
    for (int i = 0; i < 32; ++i) dst[tid + i * 256] = s4[tid + i * 256];
  }

  // softmax: 8 threads per row, 128 cols each
  const int row = tid >> 3, seg = tid & 7;
  float* Srow = S + row * 1024 + seg * 128;
  float mx = -3.402823466e38f;
  for (int c = 0; c < 128; ++c) mx = fmaxf(mx, Srow[c]);
  pmax[row * 8 + seg] = mx;
  __syncthreads();
  float rmx = pmax[row * 8];
#pragma unroll
  for (int j = 1; j < 8; ++j) rmx = fmaxf(rmx, pmax[row * 8 + j]);
  float sm = 0.0f;
  for (int c = 0; c < 128; ++c) {
    float e = __expf(Srow[c] - rmx);
    Srow[c] = e;
    sm += e;
  }
  psum[row * 8 + seg] = sm;
  __syncthreads();
  float rs = 0.0f;
#pragma unroll
  for (int j = 0; j < 8; ++j) rs += psum[row * 8 + j];
  float inv = 1.0f / rs;
  for (int c = 0; c < 128; ++c) Srow[c] *= inv;
  __syncthreads();

  // dump normalized weights (coalesced 16B)
  {
    uint4* dst = (uint4*)(weights + ((size_t)bh * N_SEQ + m0) * N_SEQ);
    const uint4* s4 = (const uint4*)S;
#pragma unroll
    for (int i = 0; i < 32; ++i) dst[tid + i * 256] = s4[tid + i * 256];
  }
}

// ---------------------------------------------------------------------------
// Kernel 3: ctx = weights @ v per (b,h). M=1024, K=1024, N=64.
// WG = 128-row strip; 8 waves, each 16 rows x 64 cols (1x4 WMMA tiles), BK=32.
// P tile is fp32->bf16 (VALU path); V tile is bf16 -> async DMA.
// ctx written bf16 in merged layout [b, s, h*64+dk].
// ---------------------------------------------------------------------------
__global__ __launch_bounds__(256)
void mha_pv_kernel(const float* __restrict__ weights, const __bf16* __restrict__ v_ws,
                   __bf16* __restrict__ ctx) {
  __shared__ __bf16 Ps[128 * 40];
  __shared__ __bf16 Vs[32 * 72];

  const int tid = threadIdx.x;
  const int lane = tid & 31, lrow = lane & 15, lhalf = lane >> 4;
  const int wid = tid >> 5;
  const int bh = blockIdx.y;
  const int m0 = blockIdx.x * 128;

  v8f acc[4] = {};

  for (int kb = 0; kb < N_SEQ; kb += 32) {
    // stage V 32x64 bf16: 256 x 16B async copies (1 per thread)
    {
      const __bf16* src = v_ws + ((size_t)bh * N_SEQ + kb) * N_DK;
      int row = tid >> 3, c4 = tid & 7;
      async_load_lds_b128(src + row * N_DK + c4 * 8, &Vs[row * 72 + c4 * 8]);
    }
    // stage P 128x32 fp32 -> bf16
#pragma unroll
    for (int i = 0; i < 8; ++i) {
      int p = tid + i * 256;
      int row = p >> 4, cp = p & 15;
      float2 g = *(const float2*)(weights + ((size_t)bh * N_SEQ + m0 + row) * N_SEQ +
                                  kb + cp * 2);
      store_pair_bf16(&Ps[row * 40 + cp * 2], g.x, g.y);
    }
    wait_async_all();
    __syncthreads();

    v16bf a = load_a_frag(Ps, 40, wid * 16, 0, lrow, lhalf);
#pragma unroll
    for (int nt = 0; nt < 4; ++nt) {
      v16bf bb = load_b_frag(Vs, 72, 0, nt * 16, lane);
      acc[nt] = wmma_bf16(a, bb, acc[nt]);
    }
    __syncthreads();
  }

  const int b = bh >> 4, h = bh & 15;
#pragma unroll
  for (int nt = 0; nt < 4; ++nt)
#pragma unroll
    for (int r = 0; r < 8; ++r) {
      int s = m0 + wid * 16 + r + lhalf * 8;
      int dk = nt * 16 + lrow;
      ctx[((size_t)b * N_SEQ + s) * N_DM + h * N_DK + dk] = (__bf16)acc[nt][r];
    }
}

// ---------------------------------------------------------------------------
// Kernel 4: output = ctx(bf16) @ Wo + bias -> fp32 d_out. Same tiling as K1.
// A tile is bf16 -> async DMA; B tile fp32->bf16 VALU path.
// ---------------------------------------------------------------------------
__global__ __launch_bounds__(256)
void mha_out_proj_kernel(const __bf16* __restrict__ Actx, const float* __restrict__ W,
                         const float* __restrict__ bias, float* __restrict__ out) {
  __shared__ __bf16 As[128 * 40];
  __shared__ __bf16 Bs[32 * 136];

  const int tid = threadIdx.x;
  const int lane = tid & 31, lrow = lane & 15, lhalf = lane >> 4;
  const int wid = tid >> 5, wm = wid >> 2, wn = wid & 3;
  const int m0 = blockIdx.x * 128, n0 = blockIdx.y * 128;

  v8f acc[4][2] = {};

  for (int kb = 0; kb < N_DM; kb += 32) {
    // stage A 128x32 bf16: 512 x 16B async copies (2 per thread)
#pragma unroll
    for (int i = 0; i < 2; ++i) {
      int p = tid + i * 256;
      int row = p >> 2, c4 = p & 3;
      async_load_lds_b128(Actx + (size_t)(m0 + row) * N_DM + kb + c4 * 8,
                          &As[row * 40 + c4 * 8]);
    }
    // stage B 32x128 fp32 -> bf16
#pragma unroll
    for (int i = 0; i < 8; ++i) {
      int p = tid + i * 256;
      int row = p >> 6, cp = p & 63;
      float2 g = *(const float2*)(W + (size_t)(kb + row) * N_DM + n0 + cp * 2);
      store_pair_bf16(&Bs[row * 136 + cp * 2], g.x, g.y);
    }
    wait_async_all();
    __syncthreads();

    v16bf af[4], bfr[2];
#pragma unroll
    for (int mt = 0; mt < 4; ++mt)
      af[mt] = load_a_frag(As, 40, wm * 64 + mt * 16, 0, lrow, lhalf);
#pragma unroll
    for (int nt = 0; nt < 2; ++nt)
      bfr[nt] = load_b_frag(Bs, 136, 0, wn * 32 + nt * 16, lane);
#pragma unroll
    for (int mt = 0; mt < 4; ++mt)
#pragma unroll
      for (int nt = 0; nt < 2; ++nt)
        acc[mt][nt] = wmma_bf16(af[mt], bfr[nt], acc[mt][nt]);
    __syncthreads();
  }

#pragma unroll
  for (int mt = 0; mt < 4; ++mt)
#pragma unroll
    for (int nt = 0; nt < 2; ++nt)
#pragma unroll
      for (int r = 0; r < 8; ++r) {
        int m = m0 + wm * 64 + mt * 16 + r + lhalf * 8;
        int n = n0 + wn * 32 + nt * 16 + lrow;
        out[(size_t)m * N_DM + n] = acc[mt][nt][r] + bias[n];
      }
}

// ---------------------------------------------------------------------------
extern "C" void kernel_launch(void* const* d_in, const int* in_sizes, int n_in,
                              void* d_out, int out_size, void* d_ws, size_t ws_size,
                              hipStream_t stream) {
  const float* Q   = (const float*)d_in[0];
  const float* K   = (const float*)d_in[1];
  const float* V   = (const float*)d_in[2];
  const float* WQw = (const float*)d_in[3];
  const float* WQb = (const float*)d_in[4];
  const float* WKw = (const float*)d_in[5];
  const float* WKb = (const float*)d_in[6];
  const float* WVw = (const float*)d_in[7];
  const float* WVb = (const float*)d_in[8];
  const float* Wow = (const float*)d_in[9];
  const float* Wob = (const float*)d_in[10];

  float* out     = (float*)d_out;
  float* weights = out + (size_t)N_BS * N_SEQ * N_DM;                 // 8M
  float* scores  = weights + (size_t)N_BS * N_HEADS * N_SEQ * N_SEQ;  // +128M

  char* ws = (char*)d_ws;
  __bf16* q_ws   = (__bf16*)(ws);                       // 16 MiB
  __bf16* kT_ws  = (__bf16*)(ws + ((size_t)16 << 20));  // 16 MiB
  __bf16* v_ws   = (__bf16*)(ws + ((size_t)32 << 20));  // 16 MiB
  __bf16* ctx_ws = (__bf16*)(ws + ((size_t)48 << 20));  // 16 MiB

  dim3 blk(256);

  // 1) projections (K transposed for the QK^T B-operand)
  mha_proj_qkv_kernel<<<dim3(64, 8), blk, 0, stream>>>(Q, WQw, WQb, q_ws, 0);
  mha_proj_qkv_kernel<<<dim3(64, 8), blk, 0, stream>>>(K, WKw, WKb, kT_ws, 1);
  mha_proj_qkv_kernel<<<dim3(64, 8), blk, 0, stream>>>(V, WVw, WVb, v_ws, 0);

  // 2) scores + softmax (dynamic LDS ~264 KB — needs CDNA5's 320KB WGP LDS)
  size_t smem = (size_t)32 * 1024 * 4      // S strip fp32
              + (size_t)64 * 1032 * 2      // kT tile bf16 (padded)
              + (size_t)32 * 72 * 2        // q tile bf16 (padded)
              + (size_t)2 * 32 * 8 * 4;    // pmax + psum
  mha_attn_softmax_kernel<<<dim3(32, 128), blk, smem, stream>>>(q_ws, kT_ws,
                                                                scores, weights);

  // 3) ctx = weights @ v
  mha_pv_kernel<<<dim3(8, 128), blk, 0, stream>>>(weights, v_ws, ctx_ws);

  // 4) output projection
  mha_out_proj_kernel<<<dim3(64, 8), blk, 0, stream>>>(ctx_ws, Wow, Wob, out);
}
